// GAT_70119636075234
// MI455X (gfx1250) — matrix operand order repaired and checked by reading, not compiled, not run on hardware
//
#include <hip/hip_runtime.h>

typedef _Float16 half_t;
typedef __attribute__((ext_vector_type(16))) _Float16 v16h;
typedef __attribute__((ext_vector_type(8)))  float    v8f;

#define SLOPE_LRELU 0.01f
#define SLOPE_ATT   0.2f
#define BN_EPS      1e-5f

union Frag { uint4 u[2]; v16h v; };

// ------------------------- small helpers -------------------------
__device__ __forceinline__ unsigned fkey(float f) {
  unsigned u = __float_as_uint(f);
  return (u & 0x80000000u) ? ~u : (u | 0x80000000u);   // monotone float->uint
}
__device__ __forceinline__ float fdec(unsigned k) {
  unsigned u = (k & 0x80000000u) ? (k & 0x7fffffffu) : ~k;
  return __uint_as_float(u);
}

__global__ void k_zero32(unsigned* p, long n) {
  long i = (long)blockIdx.x * blockDim.x + threadIdx.x;
  if (i < n) p[i] = 0u;
}

__global__ void k_cast_h(const float* __restrict__ s, half_t* __restrict__ d, long n) {
  long i = (long)blockIdx.x * blockDim.x + threadIdx.x;
  if (i < n) d[i] = (half_t)s[i];
}

// W is [K, Nc] row-major fp32 -> Wt is [NcPad, K] row-major f16 (B^T layout)
__global__ void k_wT_h(const float* __restrict__ W, half_t* __restrict__ Wt,
                       int K, int Nc, int NcPad) {
  long i = (long)blockIdx.x * blockDim.x + threadIdx.x;
  long total = (long)NcPad * K;
  if (i >= total) return;
  int n = (int)(i / K), k = (int)(i % K);
  Wt[i] = (n < Nc) ? (half_t)W[(size_t)k * Nc + n] : (half_t)0.0f;
}

__global__ void k_pad_bias(const float* __restrict__ b, float* __restrict__ bp,
                           int n, int npad) {
  int i = blockIdx.x * blockDim.x + threadIdx.x;
  if (i < npad) bp[i] = (i < n) ? b[i] : 0.0f;
}

// ------------------------- WMMA GEMM -------------------------
// C[M,Nc] = act(A[M,K](f16, lda) * Bt[Nc,K]^T(f16) + bias)
// M % 16 == 0, K % 32 == 0, Nc % 16 == 0. One 16x16 tile per wave32.
__global__ void k_gemm_wmma(const half_t* __restrict__ A, int lda,
                            const half_t* __restrict__ Bt,
                            const float* __restrict__ bias,
                            float* __restrict__ C,
                            int M, int K, int Nc, int act) {
  int lane = threadIdx.x & 31;
  int wv   = threadIdx.x >> 5;
  int tilesN = Nc >> 4;
  long totalTiles = (long)(M >> 4) * tilesN;
  long tile = (long)blockIdx.x * 8 + wv;
  if (tile >= totalTiles) return;                    // uniform per wave
  int tm = (int)(tile / tilesN);
  int tn = (int)(tile % tilesN);

  const half_t* Arow = A  + (size_t)(tm * 16 + (lane & 15)) * lda;
  const half_t* Brow = Bt + (size_t)(tn * 16 + (lane & 15)) * K;
  int ka = (lane >> 4) << 3;   // 0 or 8   (A half-lane K base)
  int kb = (lane >> 4) << 3;   // 0 or 8   (B half-lane K base)

  v8f acc = {};
  for (int k0 = 0; k0 < K; k0 += 32) {
    Frag fa, fb;
    fa.u[0] = *(const uint4*)(Arow + k0 + ka);
    fa.u[1] = *(const uint4*)(Arow + k0 + ka + 16);
    fb.u[0] = *(const uint4*)(Brow + k0 + kb);
    fb.u[1] = *(const uint4*)(Brow + k0 + kb + 16);
    acc = __builtin_amdgcn_wmma_f32_16x16x32_f16(
        false, fa.v, false, fb.v, (short)0, acc, false, false);
  }

  int col   = tn * 16 + (lane & 15);
  int mbase = tm * 16 + ((lane >> 4) << 3);
  float bv = bias ? bias[col] : 0.0f;
#pragma unroll
  for (int r = 0; r < 8; ++r) {
    float v = acc[r] + bv;
    if (act) v = (v >= 0.0f) ? v : SLOPE_LRELU * v;
    C[(size_t)(mbase + r) * Nc + col] = v;
  }
}

// ------------------------- attention coefficients -------------------------
// t: [N, Hh*D]; a_s,a_d: [Hh, D] -> as_out/ad_out: [N, Hh]
__global__ void k_alpha(const float* __restrict__ t,
                        const float* __restrict__ a_s, const float* __restrict__ a_d,
                        float* __restrict__ as_out, float* __restrict__ ad_out,
                        int N, int Hh, int D) {
  long i = (long)blockIdx.x * blockDim.x + threadIdx.x;
  if (i >= (long)N * Hh) return;
  int n = (int)(i / Hh), h = (int)(i % Hh);
  const float* p  = t + (size_t)n * Hh * D + (size_t)h * D;
  const float* ws = a_s + (size_t)h * D;
  const float* wd = a_d + (size_t)h * D;
  float ss = 0.0f, sd = 0.0f;
  for (int d = 0; d < D; ++d) { float v = p[d]; ss += v * ws[d]; sd += v * wd[d]; }
  as_out[i] = ss; ad_out[i] = sd;
}

// ------------------------- edge kernels -------------------------
__global__ void k_edge_logits(const int* __restrict__ src, const int* __restrict__ dst,
                              const float* __restrict__ as, const float* __restrict__ ad,
                              float* __restrict__ ebuf, unsigned* __restrict__ nmax,
                              long E, int Hh) {
  long i = (long)blockIdx.x * blockDim.x + threadIdx.x;
  if (i >= E * Hh) return;
  long e = i / Hh; int h = (int)(i % Hh);
  int s = src[e], d = dst[e];
  float v = as[(size_t)s * Hh + h] + ad[(size_t)d * Hh + h];
  v = (v >= 0.0f) ? v : SLOPE_ATT * v;
  ebuf[i] = v;
  atomicMax(&nmax[(size_t)d * Hh + h], fkey(v));
}

__global__ void k_edge_expsum(const int* __restrict__ dst,
                              float* __restrict__ ebuf,
                              const unsigned* __restrict__ nmax,
                              float* __restrict__ den, long E, int Hh) {
  long i = (long)blockIdx.x * blockDim.x + threadIdx.x;
  if (i >= E * Hh) return;
  long e = i / Hh; int h = (int)(i % Hh);
  int d = dst[e];
  float m = fdec(nmax[(size_t)d * Hh + h]);
  float w = __expf(ebuf[i] - m);
  ebuf[i] = w;
  unsafeAtomicAdd(&den[(size_t)d * Hh + h], w);
}

// one thread per (edge, head, dim): coalesced gather + native fp32 atomic scatter
__global__ void k_edge_scatter(const int* __restrict__ src, const int* __restrict__ dst,
                               const float* __restrict__ w, const float* __restrict__ den,
                               const float* __restrict__ t, float* __restrict__ agg,
                               long total, int Hh, int D) {
  long gid = (long)blockIdx.x * blockDim.x + threadIdx.x;
  if (gid >= total) return;
  int dloc = (int)(gid % D);
  long eh  = gid / D;
  long e   = eh / Hh; int h = (int)(eh % Hh);
  int s = src[e], dd = dst[e];
  float a = w[eh] / (den[(size_t)dd * Hh + h] + 1e-16f);
  float m = t[(size_t)s * Hh * D + (size_t)h * D + dloc] * a;
  unsafeAtomicAdd(&agg[(size_t)dd * Hh * D + (size_t)h * D + dloc], m);
}

__global__ void k_bias_lrelu(float* __restrict__ x, const float* __restrict__ b,
                             long total, int C) {
  long i = (long)blockIdx.x * blockDim.x + threadIdx.x;
  if (i >= total) return;
  float v = x[i] + b[(int)(i % C)];
  x[i] = (v >= 0.0f) ? v : SLOPE_LRELU * v;
}

// ------------------------- batchnorm -------------------------
#define BN_ROWS 64
__global__ void k_bn_stats(const float* __restrict__ g, float* __restrict__ sum,
                           float* __restrict__ sq, int N, int C) {
  __shared__ float lds[1024];           // 2 * C, C <= 512
  for (int i = threadIdx.x; i < 2 * C; i += blockDim.x) lds[i] = 0.0f;
  __syncthreads();
  int r0 = blockIdx.x * BN_ROWS;
  for (int idx = threadIdx.x; idx < BN_ROWS * C; idx += blockDim.x) {
    int r = r0 + idx / C, c = idx % C;
    if (r < N) {
      float v = g[(size_t)r * C + c];
      atomicAdd(&lds[c], v);
      atomicAdd(&lds[C + c], v * v);
    }
  }
  __syncthreads();
  for (int c = threadIdx.x; c < C; c += blockDim.x) {
    unsafeAtomicAdd(&sum[c], lds[c]);
    unsafeAtomicAdd(&sq[c], lds[C + c]);
  }
}

__global__ void k_bn_finalize(const float* __restrict__ sum, const float* __restrict__ sq,
                              const float* __restrict__ gamma, const float* __restrict__ beta,
                              float* __restrict__ scale, float* __restrict__ shift,
                              int C, float invN) {
  int c = blockIdx.x * blockDim.x + threadIdx.x;
  if (c >= C) return;
  float mu  = sum[c] * invN;
  float var = sq[c] * invN - mu * mu;
  float sc  = gamma[c] * rsqrtf(var + BN_EPS);
  scale[c] = sc;
  shift[c] = beta[c] - mu * sc;
}

__global__ void k_bn_apply_h(const float* __restrict__ g,
                             const float* __restrict__ scale, const float* __restrict__ shift,
                             half_t* __restrict__ dst, int ldd, int N, int C) {
  long i = (long)blockIdx.x * blockDim.x + threadIdx.x;
  if (i >= (long)N * C) return;
  int r = (int)(i / C), c = (int)(i % C);
  dst[(size_t)r * ldd + c] = (half_t)(g[i] * scale[c] + shift[c]);
}

// ------------------------- log-softmax -------------------------
__global__ void k_logsoftmax(const float* __restrict__ z, float* __restrict__ out,
                             int N, int ldz, int C) {
  int r = blockIdx.x * blockDim.x + threadIdx.x;
  if (r >= N) return;
  const float* p = z + (size_t)r * ldz;
  float m = -3.0e38f;
  for (int c = 0; c < C; ++c) m = fmaxf(m, p[c]);
  float s = 0.0f;
  for (int c = 0; c < C; ++c) s += __expf(p[c] - m);
  float l = m + __logf(s);
  for (int c = 0; c < C; ++c) out[(size_t)r * C + c] = p[c] - l;
}

// ------------------------- orchestration -------------------------
static inline long cdiv(long a, long b) { return (a + b - 1) / b; }

extern "C" void kernel_launch(void* const* d_in, const int* in_sizes, int n_in,
                              void* d_out, int out_size, void* d_ws, size_t ws_size,
                              hipStream_t stream) {
  const int F_IN = 256, H = 128, HEADS = 4, C_OUT = 40, C_PAD = 48, CAT = 768;
  const int N = in_sizes[0] / F_IN;
  const long E = in_sizes[1] / 2;

  const float* x      = (const float*)d_in[0];
  const int*   src    = (const int*)d_in[1];
  const int*   dst    = (const int*)d_in[1] + E;
  const float* fw_W   = (const float*)d_in[2];
  const float* fw_b   = (const float*)d_in[3];
  const float* c1_W   = (const float*)d_in[4];
  const float* c1_as  = (const float*)d_in[5];
  const float* c1_ad  = (const float*)d_in[6];
  const float* c1_b   = (const float*)d_in[7];
  const float* c2_W   = (const float*)d_in[8];
  const float* c2_as  = (const float*)d_in[9];
  const float* c2_ad  = (const float*)d_in[10];
  const float* c2_b   = (const float*)d_in[11];
  const float* bn1_g  = (const float*)d_in[12];
  const float* bn1_b  = (const float*)d_in[13];
  const float* bn2_g  = (const float*)d_in[14];
  const float* bn2_b  = (const float*)d_in[15];
  const float* bn3_g  = (const float*)d_in[16];
  const float* bn3_b  = (const float*)d_in[17];
  const float* l2_W   = (const float*)d_in[18];
  const float* l2_b   = (const float*)d_in[19];
  float* out = (float*)d_out;

  // workspace layout (256B aligned bump allocator)
  char* base = (char*)d_ws;
  size_t cur = 0;
  auto alloc = [&](size_t bytes) -> char* {
    char* p = base + cur;
    cur = (cur + bytes + 255) & ~(size_t)255;
    return p;
  };
  half_t* xh    = (half_t*)alloc((size_t)N * F_IN * 2);
  half_t* wfwT  = (half_t*)alloc((size_t)H * F_IN * 2);
  half_t* wc1T  = (half_t*)alloc((size_t)(HEADS * H) * H * 2);
  half_t* wc2T  = (half_t*)alloc((size_t)H * (HEADS * H) * 2);
  half_t* wl2T  = (half_t*)alloc((size_t)C_PAD * CAT * 2);
  float*  l2bp  = (float*)alloc(C_PAD * 4);
  float*  bsum  = (float*)alloc(512 * 4);
  float*  bsq   = (float*)alloc(512 * 4);
  float*  bscale= (float*)alloc(512 * 4);
  float*  bshift= (float*)alloc(512 * 4);
  half_t* catH  = (half_t*)alloc((size_t)N * CAT * 2);     // [h0 | h1 | h2] f16
  float*  g0    = (float*)alloc((size_t)N * H * 4);        // stage1 out; reused as t2
  float*  t1    = (float*)alloc((size_t)N * HEADS * H * 4);// conv1 h; reused agg2+zbuf
  float*  agg1  = (float*)alloc((size_t)N * HEADS * H * 4);
  float*  asb   = (float*)alloc((size_t)N * HEADS * 4);
  float*  adb   = (float*)alloc((size_t)N * HEADS * 4);
  unsigned* nmax= (unsigned*)alloc((size_t)N * HEADS * 4);
  float*  den   = (float*)alloc((size_t)N * HEADS * 4);
  float*  ebuf  = (float*)alloc((size_t)E * HEADS * 4);
  float*  t2    = g0;                 // reuse (stage1 result lives in catH by then)
  float*  agg2  = t1;                 // reuse (t1 dead after conv1 scatter)
  float*  zbuf  = t1 + (size_t)N * H; // [N,48], fits inside t1 region

  const int TB = 256;
  float invN = 1.0f / (float)N;

  // ---- prep: casts + weight transposes ----
  k_cast_h<<<cdiv((long)N * F_IN, TB), TB, 0, stream>>>(x, xh, (long)N * F_IN);
  k_wT_h<<<cdiv((long)H * F_IN, TB), TB, 0, stream>>>(fw_W, wfwT, F_IN, H, H);
  k_wT_h<<<cdiv((long)(HEADS * H) * H, TB), TB, 0, stream>>>(c1_W, wc1T, H, HEADS * H, HEADS * H);
  k_wT_h<<<cdiv((long)H * (HEADS * H), TB), TB, 0, stream>>>(c2_W, wc2T, HEADS * H, H, H);
  k_wT_h<<<cdiv((long)C_PAD * CAT, TB), TB, 0, stream>>>(l2_W, wl2T, CAT, C_OUT, C_PAD);
  k_pad_bias<<<1, 64, 0, stream>>>(l2_b, l2bp, C_OUT, C_PAD);

  // ---- stage 1: h0 = BN1(lrelu(x @ fw_W + fw_b)) -> catH[:, 0:128] ----
  {
    long tiles = (long)(N / 16) * (H / 16);
    k_gemm_wmma<<<cdiv(tiles, 8), TB, 0, stream>>>(xh, F_IN, wfwT, fw_b, g0, N, F_IN, H, 1);
    k_zero32<<<cdiv(2L * H, TB), TB, 0, stream>>>((unsigned*)bsum, H);
    k_zero32<<<cdiv(2L * H, TB), TB, 0, stream>>>((unsigned*)bsq, H);
    k_bn_stats<<<cdiv(N, BN_ROWS), TB, 0, stream>>>(g0, bsum, bsq, N, H);
    k_bn_finalize<<<cdiv(H, 64), 64, 0, stream>>>(bsum, bsq, bn1_g, bn1_b, bscale, bshift, H, invN);
    k_bn_apply_h<<<cdiv((long)N * H, TB), TB, 0, stream>>>(g0, bscale, bshift, catH + 0, CAT, N, H);
  }

  // ---- conv1: GATConv(128 -> 128, heads=4) ----
  {
    long tiles = (long)(N / 16) * ((HEADS * H) / 16);
    k_gemm_wmma<<<cdiv(tiles, 8), TB, 0, stream>>>(catH + 0, CAT, wc1T, nullptr, t1, N, H, HEADS * H, 0);
    k_alpha<<<cdiv((long)N * HEADS, TB), TB, 0, stream>>>(t1, c1_as, c1_ad, asb, adb, N, HEADS, H);
    k_zero32<<<cdiv((long)N * HEADS, TB), TB, 0, stream>>>(nmax, (long)N * HEADS);
    k_zero32<<<cdiv((long)N * HEADS, TB), TB, 0, stream>>>((unsigned*)den, (long)N * HEADS);
    k_zero32<<<cdiv((long)N * HEADS * H, TB), TB, 0, stream>>>((unsigned*)agg1, (long)N * HEADS * H);
    k_edge_logits<<<cdiv(E * HEADS, TB), TB, 0, stream>>>(src, dst, asb, adb, ebuf, nmax, E, HEADS);
    k_edge_expsum<<<cdiv(E * HEADS, TB), TB, 0, stream>>>(dst, ebuf, nmax, den, E, HEADS);
    long total = E * HEADS * H;
    k_edge_scatter<<<cdiv(total, TB), TB, 0, stream>>>(src, dst, ebuf, den, t1, agg1, total, HEADS, H);
    k_bias_lrelu<<<cdiv((long)N * HEADS * H, TB), TB, 0, stream>>>(agg1, c1_b, (long)N * HEADS * H, HEADS * H);
    int C1 = HEADS * H;
    k_zero32<<<cdiv((long)C1, 64L), 64, 0, stream>>>((unsigned*)bsum, C1);
    k_zero32<<<cdiv((long)C1, 64L), 64, 0, stream>>>((unsigned*)bsq, C1);
    k_bn_stats<<<cdiv(N, BN_ROWS), TB, 0, stream>>>(agg1, bsum, bsq, N, C1);
    k_bn_finalize<<<cdiv(C1, 64), 64, 0, stream>>>(bsum, bsq, bn2_g, bn2_b, bscale, bshift, C1, invN);
    k_bn_apply_h<<<cdiv((long)N * C1, TB), TB, 0, stream>>>(agg1, bscale, bshift, catH + H, CAT, N, C1);
  }

  // ---- conv2: GATConv(512 -> 128, heads=1) ----
  {
    long tiles = (long)(N / 16) * (H / 16);
    k_gemm_wmma<<<cdiv(tiles, 8), TB, 0, stream>>>(catH + H, CAT, wc2T, nullptr, t2, N, HEADS * H, H, 0);
    k_alpha<<<cdiv((long)N, TB), TB, 0, stream>>>(t2, c2_as, c2_ad, asb, adb, N, 1, H);
    k_zero32<<<cdiv((long)N, TB), TB, 0, stream>>>(nmax, (long)N);
    k_zero32<<<cdiv((long)N, TB), TB, 0, stream>>>((unsigned*)den, (long)N);
    k_zero32<<<cdiv((long)N * H, TB), TB, 0, stream>>>((unsigned*)agg2, (long)N * H);
    k_edge_logits<<<cdiv(E, TB), TB, 0, stream>>>(src, dst, asb, adb, ebuf, nmax, E, 1);
    k_edge_expsum<<<cdiv(E, TB), TB, 0, stream>>>(dst, ebuf, nmax, den, E, 1);
    long total = E * H;
    k_edge_scatter<<<cdiv(total, TB), TB, 0, stream>>>(src, dst, ebuf, den, t2, agg2, total, 1, H);
    k_bias_lrelu<<<cdiv((long)N * H, TB), TB, 0, stream>>>(agg2, c2_b, (long)N * H, H);
    k_zero32<<<cdiv((long)H, 64L), 64, 0, stream>>>((unsigned*)bsum, H);
    k_zero32<<<cdiv((long)H, 64L), 64, 0, stream>>>((unsigned*)bsq, H);
    k_bn_stats<<<cdiv(N, BN_ROWS), TB, 0, stream>>>(agg2, bsum, bsq, N, H);
    k_bn_finalize<<<cdiv(H, 64), 64, 0, stream>>>(bsum, bsq, bn3_g, bn3_b, bscale, bshift, H, invN);
    k_bn_apply_h<<<cdiv((long)N * H, TB), TB, 0, stream>>>(agg2, bscale, bshift, catH + H + HEADS * H, CAT, N, H);
  }

  // ---- lin2 on concatenated features + log_softmax ----
  {
    long tiles = (long)(N / 16) * (C_PAD / 16);
    k_gemm_wmma<<<cdiv(tiles, 8), TB, 0, stream>>>(catH, CAT, wl2T, l2bp, zbuf, N, CAT, C_PAD, 0);
    k_logsoftmax<<<cdiv(N, TB), TB, 0, stream>>>(zbuf, out, N, C_PAD, C_OUT);
  }
}
